// net_68908455297444
// MI455X (gfx1250) — matrix operand order repaired and checked by reading, not compiled, not run on hardware
//
#include <hip/hip_runtime.h>
#include <math.h>

#define N_NODES 50000
#define E0      800000          // raw edges
#define ET      850000          // edges + self loops
#define KDIM    128
#define HEADS   4
#define N_CLS   47

typedef __attribute__((ext_vector_type(16))) __bf16 v16bf;
typedef __attribute__((ext_vector_type(8)))  __bf16 v8bf;
typedef __attribute__((ext_vector_type(8)))  float  v8f;

// ---------- native fp32 atomic add (global_atomic_add_f32, no CAS loop) ----------
__device__ __forceinline__ void atomic_add_f32(float* p, float v) {
  unsafeAtomicAdd(p, v);
}

// ---------- monotonic uint encoding of float for atomic max ----------
__device__ __forceinline__ unsigned fkey(float f) {
  unsigned u = __float_as_uint(f);
  return (u & 0x80000000u) ? ~u : (u | 0x80000000u);
}
__device__ __forceinline__ float fdec(unsigned k) {
  return __uint_as_float((k & 0x80000000u) ? (k & 0x7FFFFFFFu) : ~k);
}
#define NEG_INF_KEY 0x007FFFFFu   // fkey(-inf)

// ---------- edge endpoint helpers (self loops appended arithmetically) ----------
__device__ __forceinline__ int e_src(const int* __restrict__ ei, int e) {
  return (e < E0) ? ei[e] : (e - E0);
}
__device__ __forceinline__ int e_dst(const int* __restrict__ ei, int e) {
  return (e < E0) ? ei[E0 + e] : (e - E0);
}

// =====================================================================
// Precision prep: fp32 -> bf16 activation copy; weight transpose+pad.
// =====================================================================
__global__ void f32_to_bf16(const float* __restrict__ src, __bf16* __restrict__ dst, size_t n)
{
  size_t i = (size_t)blockIdx.x * blockDim.x + threadIdx.x;
  if (i < n) dst[i] = (__bf16)src[i];
}

// W [K,N] row-major f32  ->  Wt [Npad,K] bf16 (column-major, zero padded rows)
__global__ void w_transpose_bf16(const float* __restrict__ W, int N, int Npad,
                                 __bf16* __restrict__ Wt)
{
  int idx = blockIdx.x * blockDim.x + threadIdx.x;   // over Npad*K
  if (idx >= Npad * KDIM) return;
  int n = idx / KDIM, k = idx % KDIM;
  Wt[idx] = (n < N) ? (__bf16)W[(size_t)k * N + n] : (__bf16)0.0f;
}

// =====================================================================
// WMMA bf16 GEMM: Hout[M,Nreal] = Xb[M,128] * Wt^T   (guard-free loads)
// One wave per 16x16 tile; wave w of the block owns column tile w.
//   A frag (ISA 7.12.2): lane<16 row m, K {0-7,16-23}; lane>=16 K {8-15,24-31}
//   B frag: lane = column, same K split as A: lane<16 K {0-7,16-23}; lane>=16 K {8-15,24-31}
// =====================================================================
__global__ void gemm_wmma(const __bf16* __restrict__ Xb,
                          const __bf16* __restrict__ Wt,
                          float* __restrict__ Hout, int ldh, int Nreal)
{
  const int wave  = threadIdx.x >> 5;
  const int lane  = threadIdx.x & 31;
  const int row0  = blockIdx.x * 16;
  const int col0  = wave * 16;
  const int mlane = lane & 15;
  const int hi    = lane >> 4;

  const __bf16* arow = Xb + (size_t)(row0 + mlane) * KDIM;
  const __bf16* brow = Wt + (size_t)(col0 + mlane) * KDIM;

  v8f acc = {};
#pragma unroll
  for (int k0 = 0; k0 < KDIM; k0 += 32) {
    v8bf a0 = *(const v8bf*)(arow + k0 + hi * 8);        // 16B aligned
    v8bf a1 = *(const v8bf*)(arow + k0 + 16 + hi * 8);
    v8bf b0 = *(const v8bf*)(brow + k0 + hi * 8);
    v8bf b1 = *(const v8bf*)(brow + k0 + 16 + hi * 8);
    v16bf a = __builtin_shufflevector(a0, a1, 0,1,2,3,4,5,6,7,8,9,10,11,12,13,14,15);
    v16bf b = __builtin_shufflevector(b0, b1, 0,1,2,3,4,5,6,7,8,9,10,11,12,13,14,15);
    acc = __builtin_amdgcn_wmma_f32_16x16x32_bf16(
        /*neg_a=*/false, a, /*neg_b=*/false, b,
        /*c_mod=*/(short)0, acc, /*reuse_a=*/false, /*reuse_b=*/false);
  }

  const int mb = hi * 8;
  const int c  = col0 + mlane;
  if (c < Nreal) {
#pragma unroll
    for (int i = 0; i < 8; ++i)
      Hout[(size_t)(row0 + mb + i) * ldh + c] = acc[i];
  }
}

// =====================================================================
// Per-(node, head) attention coefficients: a_src/a_dst = <h[n,h,:], att>
// =====================================================================
__global__ void attn_coef(const float* __restrict__ Hm, int ldh,
                          const float* __restrict__ as_, const float* __restrict__ ad_,
                          float* __restrict__ Asrc, float* __restrict__ Adst, int C)
{
  int idx = blockIdx.x * blockDim.x + threadIdx.x;   // n*HEADS + h
  if (idx >= N_NODES * HEADS) return;
  int n = idx >> 2, h = idx & 3;
  const float* row = Hm + (size_t)n * ldh + h * C;
  float s = 0.f, d = 0.f;
  for (int c = 0; c < C; ++c) {
    float v = row[c];
    s += v * as_[h * C + c];
    d += v * ad_[h * C + c];
  }
  Asrc[idx] = s;
  Adst[idx] = d;
}

// ---------- fills ----------
__global__ void fill_f32(float* p, size_t n, float v) {
  size_t i = (size_t)blockIdx.x * blockDim.x + threadIdx.x;
  if (i < n) p[i] = v;
}
__global__ void fill_u32(unsigned* p, size_t n, unsigned v) {
  size_t i = (size_t)blockIdx.x * blockDim.x + threadIdx.x;
  if (i < n) p[i] = v;
}

// ---------- edge pass 1: alpha = leakyrelu(asrc[s]+adst[d]); segment max ----------
__global__ void edge_alpha_max(const int* __restrict__ ei,
                               const float* __restrict__ Asrc, const float* __restrict__ Adst,
                               float* __restrict__ ebuf, unsigned* __restrict__ mbuf)
{
  int idx = blockIdx.x * blockDim.x + threadIdx.x;   // e*HEADS + h
  if (idx >= ET * HEADS) return;
  int e = idx >> 2, h = idx & 3;
  int s = e_src(ei, e), d = e_dst(ei, e);
  float a = Asrc[s * HEADS + h] + Adst[d * HEADS + h];
  a = (a >= 0.f) ? a : 0.2f * a;
  ebuf[idx] = a;
  atomicMax(&mbuf[d * HEADS + h], fkey(a));
}

// ---------- edge pass 2: e = exp(alpha - max[dst]); segment sum ----------
__global__ void edge_exp_sum(const int* __restrict__ ei,
                             float* __restrict__ ebuf,
                             const unsigned* __restrict__ mbuf,
                             float* __restrict__ den)
{
  int idx = blockIdx.x * blockDim.x + threadIdx.x;
  if (idx >= ET * HEADS) return;
  int e = idx >> 2, h = idx & 3;
  int d = e_dst(ei, e);
  float ex = __expf(ebuf[idx] - fdec(mbuf[d * HEADS + h]));
  ebuf[idx] = ex;
  atomic_add_f32(&den[d * HEADS + h], ex);
}

// ---------- edge pass 3: out[dst] += h[src] * (e / den[dst])  (one block per edge) ----------
__global__ void edge_aggregate(const int* __restrict__ ei,
                               const float* __restrict__ Hm, int ldh,
                               const float* __restrict__ ebuf,
                               const float* __restrict__ den,
                               float* __restrict__ outb, int HC, int C)
{
  int e = blockIdx.x;
  int c = threadIdx.x;
  if (c >= HC) return;
  int s = e_src(ei, e), d = e_dst(ei, e);
  int h = c / C;
  float w = ebuf[e * HEADS + h] / den[d * HEADS + h];
  atomic_add_f32(&outb[(size_t)d * HC + c], Hm[(size_t)s * ldh + c] * w);
}

// ---------- bias + relu (layers 1,2; HC==128) ----------
__global__ void bias_relu(float* __restrict__ o, const float* __restrict__ b, size_t total)
{
  size_t i = (size_t)blockIdx.x * blockDim.x + threadIdx.x;
  if (i >= total) return;
  float v = o[i] + b[i & 127];
  o[i] = v > 0.f ? v : 0.f;
}

// ---------- layer 3 tail: mean over heads + bias + log_softmax ----------
__global__ void final_logsoftmax(const float* __restrict__ agg,
                                 const float* __restrict__ b3,
                                 float* __restrict__ out)
{
  int n = blockIdx.x * blockDim.x + threadIdx.x;
  if (n >= N_NODES) return;
  const float* r = agg + (size_t)n * (HEADS * N_CLS);
  float v[N_CLS];
  float mx = -INFINITY;
  for (int c = 0; c < N_CLS; ++c) {
    float s = r[c] + r[N_CLS + c] + r[2 * N_CLS + c] + r[3 * N_CLS + c];
    s = s * 0.25f + b3[c];
    v[c] = s;
    mx = fmaxf(mx, s);
  }
  float sum = 0.f;
  for (int c = 0; c < N_CLS; ++c) sum += __expf(v[c] - mx);
  float lse = mx + __logf(sum);
  for (int c = 0; c < N_CLS; ++c) out[(size_t)n * N_CLS + c] = v[c] - lse;
}

// =====================================================================
static void run_gat_layer(const float* X, const float* W, int Ncols, int Npad,
                          int ldh, int C, int HC,
                          const float* as_, const float* ad_, const int* ei,
                          __bf16* Xbf, __bf16* Wt, float* g,
                          float* asrc, float* adst, unsigned* mbuf,
                          float* den, float* ebuf, float* outb, hipStream_t stream)
{
  // prep bf16 operands
  const size_t xn = (size_t)N_NODES * KDIM;
  f32_to_bf16<<<(unsigned)((xn + 255) / 256), 256, 0, stream>>>(X, Xbf, xn);
  const int wn = Npad * KDIM;
  w_transpose_bf16<<<(wn + 255) / 256, 256, 0, stream>>>(W, Ncols, Npad, Wt);

  // GEMM: one wave per 16x16 tile, Npad/16 waves per block
  gemm_wmma<<<dim3(N_NODES / 16), dim3(32 * (Npad / 16)), 0, stream>>>(
      Xbf, Wt, g, ldh, Ncols);

  const int nh = N_NODES * HEADS;
  attn_coef<<<(nh + 255) / 256, 256, 0, stream>>>(g, ldh, as_, ad_, asrc, adst, C);
  fill_u32<<<(nh + 255) / 256, 256, 0, stream>>>(mbuf, (size_t)nh, NEG_INF_KEY);
  fill_f32<<<(nh + 255) / 256, 256, 0, stream>>>(den, (size_t)nh, 0.f);

  const int eh = ET * HEADS;
  edge_alpha_max<<<(eh + 255) / 256, 256, 0, stream>>>(ei, asrc, adst, ebuf, mbuf);
  edge_exp_sum<<<(eh + 255) / 256, 256, 0, stream>>>(ei, ebuf, mbuf, den);

  const size_t tot = (size_t)N_NODES * HC;
  fill_f32<<<(unsigned)((tot + 255) / 256), 256, 0, stream>>>(outb, tot, 0.f);
  const int bthreads = ((HC + 31) / 32) * 32;
  edge_aggregate<<<ET, bthreads, 0, stream>>>(ei, g, ldh, ebuf, den, outb, HC, C);
}

extern "C" void kernel_launch(void* const* d_in, const int* in_sizes, int n_in,
                              void* d_out, int out_size, void* d_ws, size_t ws_size,
                              hipStream_t stream)
{
  (void)in_sizes; (void)n_in; (void)out_size; (void)ws_size;
  const float* x   = (const float*)d_in[0];
  const int*   ei  = (const int*)d_in[1];
  const float* W1  = (const float*)d_in[2];
  const float* a1s = (const float*)d_in[3];
  const float* a1d = (const float*)d_in[4];
  const float* b1  = (const float*)d_in[5];
  const float* W2  = (const float*)d_in[6];
  const float* a2s = (const float*)d_in[7];
  const float* a2d = (const float*)d_in[8];
  const float* b2  = (const float*)d_in[9];
  const float* W3  = (const float*)d_in[10];
  const float* a3s = (const float*)d_in[11];
  const float* a3d = (const float*)d_in[12];
  const float* b3  = (const float*)d_in[13];
  float* out = (float*)d_out;

  char* ws = (char*)d_ws;
  size_t off = 0;
  auto alloc = [&](size_t bytes) { void* p = (void*)(ws + off); off += (bytes + 255) & ~(size_t)255; return p; };
  float*    g    = (float*)alloc((size_t)N_NODES * 192 * 4);   // GEMM output (reused)
  float*    x1   = (float*)alloc((size_t)N_NODES * 128 * 4);   // layer-1 output
  float*    x2   = (float*)alloc((size_t)N_NODES * 128 * 4);   // layer-2 output
  float*    agg3 = (float*)alloc((size_t)N_NODES * 188 * 4);   // layer-3 aggregate
  float*    asrc = (float*)alloc((size_t)N_NODES * HEADS * 4);
  float*    adst = (float*)alloc((size_t)N_NODES * HEADS * 4);
  unsigned* mbuf = (unsigned*)alloc((size_t)N_NODES * HEADS * 4);
  float*    den  = (float*)alloc((size_t)N_NODES * HEADS * 4);
  float*    ebuf = (float*)alloc((size_t)ET * HEADS * 4);
  __bf16*   Xbf  = (__bf16*)alloc((size_t)N_NODES * KDIM * 2); // bf16 activations (reused)
  __bf16*   Wt   = (__bf16*)alloc((size_t)192 * KDIM * 2);     // bf16 transposed weights

  // ---- layer 1: 128 -> 4x32, concat, relu ----
  run_gat_layer(x, W1, 128, 128, 128, 32, 128, a1s, a1d, ei,
                Xbf, Wt, g, asrc, adst, mbuf, den, ebuf, x1, stream);
  bias_relu<<<(N_NODES * 128 + 255) / 256, 256, 0, stream>>>(x1, b1, (size_t)N_NODES * 128);

  // ---- layer 2: 128 -> 4x32, concat, relu ----
  run_gat_layer(x1, W2, 128, 128, 128, 32, 128, a2s, a2d, ei,
                Xbf, Wt, g, asrc, adst, mbuf, den, ebuf, x2, stream);
  bias_relu<<<(N_NODES * 128 + 255) / 256, 256, 0, stream>>>(x2, b2, (size_t)N_NODES * 128);

  // ---- layer 3: 128 -> 4x47, mean over heads, bias, log_softmax ----
  run_gat_layer(x2, W3, 188, 192, 188, N_CLS, 188, a3s, a3d, ei,
                Xbf, Wt, g, asrc, adst, mbuf, den, ebuf, agg3, stream);
  final_logsoftmax<<<(N_NODES + 255) / 256, 256, 0, stream>>>(agg3, b3, out);
}
